// AttentionBlock_25520695673083
// MI455X (gfx1250) — compile-verified, matrix operand order checked by reading, not run on hardware
//
#include <hip/hip_runtime.h>
#include <hip/hip_bf16.h>

typedef __attribute__((ext_vector_type(16))) _Float16 v16h;
typedef __attribute__((ext_vector_type(8)))  _Float16 v8h;
typedef __attribute__((ext_vector_type(8)))  float    v8f;
typedef __attribute__((ext_vector_type(4)))  int      v4i;

static constexpr int BB   = 16;
static constexpr int CC   = 256;
static constexpr int DQK  = 32;    // C/8
static constexpr int NN   = 4096;  // H*W
static constexpr int PADR = 40;    // padded LDS row stride (halfwords) for 32-hw rows

// ---- CDNA5 async global->LDS copy (ASYNCcnt path), with safe fallback -----
#if __has_builtin(__builtin_amdgcn_global_load_async_to_lds_b128) && \
    __has_builtin(__builtin_amdgcn_s_wait_asynccnt)
#define GLOBAL_AS __attribute__((address_space(1)))
#define LDS_AS    __attribute__((address_space(3)))
#define ASYNC_COPY_B128(dst, src)                                             \
    __builtin_amdgcn_global_load_async_to_lds_b128(                           \
        (GLOBAL_AS v4i*)(GLOBAL_AS void*)(void*)(src),                        \
        (LDS_AS v4i*)(LDS_AS void*)(dst), 0, 0)
#define ASYNC_WAIT() __builtin_amdgcn_s_wait_asynccnt(0)
#else
#define ASYNC_COPY_B128(dst, src) (*(v8h*)(dst) = *(const v8h*)(src))
#define ASYNC_WAIT() ((void)0)
#endif

union V16U { v16h v; v8h h[2]; };

// Per-lane fragment load for a row-major [16 x 32] f16 tile (A layout; B layout
// assumed mirrored). p = base of this lane's row/column (global or LDS).
// Lanes 0-15 take K elements {0..7,16..23}; lanes 16-31 take {8..15,24..31}.
__device__ __forceinline__ v16h ld_frag(const _Float16* p, int hl) {
    V16U u;
    u.h[0] = *(const v8h*)(p + hl * 8);
    u.h[1] = *(const v8h*)(p + 16 + hl * 8);
    return u.v;
}

__device__ __forceinline__ v8f wmma_f16(v16h a, v16h b, v8f c) {
    return __builtin_amdgcn_wmma_f32_16x16x32_f16(false, a, false, b,
                                                  (short)0, c, false, false);
}

// ---------------------------------------------------------------------------
// Kernel 0: one-shot weight conversion to f16.
//   Wqkh = [Wq; Wk] (64 x 256, row-major),  Wvh = Wv (256 x 256, row-major)
// ---------------------------------------------------------------------------
__global__ __launch_bounds__(256) void wcvt_kernel(
    const float* __restrict__ Wq, const float* __restrict__ Wk,
    const float* __restrict__ Wv,
    _Float16* __restrict__ Wqkh, _Float16* __restrict__ Wvh)
{
    const int i = blockIdx.x * 256 + threadIdx.x;   // 0 .. CC*CC-1
    if (i < CC * CC) Wvh[i] = (_Float16)Wv[i];
    if (i < DQK * CC) {
        Wqkh[i]            = (_Float16)Wq[i];
        Wqkh[DQK * CC + i] = (_Float16)Wk[i];
    }
}

// ---------------------------------------------------------------------------
// Kernel 1: WMMA Q/K/V projections with async-LDS double-buffered weights.
// Each wave owns 16 tokens. One x fragment per K-step serves as BOTH
//   A of [Q;K]^T = X^T x Wqk^T (4 tiles)  and  B of V = Wv x X (16 tiles).
// Weight slab (320 rows x 32 ch = 20KB f16) staged per step, shared by 4 waves.
// ---------------------------------------------------------------------------
__device__ __forceinline__ void proj_stage(
    const _Float16* __restrict__ wqk, const _Float16* __restrict__ wv,
    _Float16* ws, int tid)
{
    #pragma unroll
    for (int i = 0; i < 10; ++i) {                 // 320 rows * 4 chunks / 128 thr
        const int idx = i * 128 + tid;
        const int row = idx >> 2, ch = (idx & 3) * 8;
        const _Float16* src = (row < 64) ? (wqk + row * CC + ch)
                                         : (wv + (size_t)(row - 64) * CC + ch);
        ASYNC_COPY_B128(ws + row * PADR + ch, src);
    }
}

__global__ __launch_bounds__(128) void proj_kernel(
    const float* __restrict__ x,
    const _Float16* __restrict__ Wqkh, const _Float16* __restrict__ Wvh,
    const float* __restrict__ bq, const float* __restrict__ bk,
    const float* __restrict__ bv,
    _Float16* __restrict__ Qh, _Float16* __restrict__ Kh, _Float16* __restrict__ Vh)
{
    __shared__ _Float16 Ws[2][320 * PADR];         // 2 x 25.6KB

    const int tid  = threadIdx.x;
    const int b    = blockIdx.x >> 6;
    const int wave = tid >> 5;
    const int lane = tid & 31;
    const int hl   = lane >> 4;
    const int l15  = lane & 15;
    const int n0   = (blockIdx.x & 63) * 64 + wave * 16;
    const int n    = n0 + l15;

    v8f Dqk[4];
    v8f Ov[16];
    #pragma unroll
    for (int t = 0; t < 4; ++t)  Dqk[t] = v8f{};
    #pragma unroll
    for (int t = 0; t < 16; ++t) Ov[t] = v8f{};

    const float* xcol = x + (size_t)b * CC * NN + n;

    proj_stage(Wqkh, Wvh, &Ws[0][0], tid);         // k0 = 0 into buffer 0

    // x fragment for k0 = 0 (register double-buffered)
    V16U Xcur;
    #pragma unroll
    for (int t = 0; t < 8; ++t) {
        Xcur.h[0][t] = (_Float16)xcol[(size_t)(hl * 8 + t) * NN];
        Xcur.h[1][t] = (_Float16)xcol[(size_t)(16 + hl * 8 + t) * NN];
    }

    for (int k0 = 0; k0 < CC; k0 += 32) {
        const int p = (k0 >> 5) & 1;
        ASYNC_WAIT();
        __syncthreads();
        if (k0 + 32 < CC)
            proj_stage(Wqkh + k0 + 32, Wvh + k0 + 32, &Ws[p ^ 1][0], tid);

        V16U Xnxt = Xcur;
        if (k0 + 32 < CC) {
            #pragma unroll
            for (int t = 0; t < 8; ++t) {
                Xnxt.h[0][t] = (_Float16)xcol[(size_t)(k0 + 32 + hl * 8 + t) * NN];
                Xnxt.h[1][t] = (_Float16)xcol[(size_t)(k0 + 48 + hl * 8 + t) * NN];
            }
        }

        const _Float16* wsp = &Ws[p][0];
        #pragma unroll
        for (int ot = 0; ot < 4; ++ot) {           // Q/K tiles: D[tok, o]
            const v16h Bw = ld_frag(wsp + (ot * 16 + l15) * PADR, hl);
            Dqk[ot] = wmma_f16(Xcur.v, Bw, Dqk[ot]);
        }
        #pragma unroll
        for (int ct = 0; ct < 16; ++ct) {          // V tiles: D[c, tok]
            const v16h Aw = ld_frag(wsp + (64 + ct * 16 + l15) * PADR, hl);
            Ov[ct] = wmma_f16(Aw, Xcur.v, Ov[ct]);
        }
        Xcur = Xnxt;
    }

    // Q/K epilogue: C-layout rows = tokens (hl*8 + t), cols o = l15
    #pragma unroll
    for (int ot = 0; ot < 4; ++ot) {
        const bool isq = (ot < 2);
        const int  oo  = (isq ? ot : (ot - 2)) * 16 + l15;   // 0..31
        const float bias = isq ? bq[oo] : bk[oo];
        _Float16* dst = (isq ? Qh : Kh) + (size_t)b * NN * DQK + oo;
        #pragma unroll
        for (int t = 0; t < 8; ++t) {
            const int tok = n0 + hl * 8 + t;
            dst[(size_t)tok * DQK] = (_Float16)(Dqk[ot][t] + bias);
        }
    }
    // V epilogue: C-layout rows = channels (ct*16 + hl*8 + t), cols tok = l15
    #pragma unroll
    for (int ct = 0; ct < 16; ++ct) {
        #pragma unroll
        for (int t = 0; t < 8; ++t) {
            const int c = ct * 16 + hl * 8 + t;
            Vh[((size_t)b * CC + c) * NN + n] = (_Float16)(Ov[ct][t] + bv[c]);
        }
    }
}

// ---------------------------------------------------------------------------
// Kernel 2: flash attention; K/V tiles async-staged to LDS, double-buffered.
// Each wave owns 16 query rows (col index i = lane&15 in all C-layouts).
// Per 32-key step:  St = K_tile x Qfrag (2 wmma) -> online softmax (per-lane
// m,l; one shfl_xor(16)) -> O^T += V_tile x P^T (16 wmma), with P^T consumed
// straight out of the St accumulator registers.
// ---------------------------------------------------------------------------
__device__ __forceinline__ void fa_stage(
    const _Float16* __restrict__ kg, const _Float16* __restrict__ vg,
    _Float16* ks, _Float16* vs, int tid)
{
    {   // K tile: 32 rows x 32 d
        const int row = tid >> 2, ch = (tid & 3) * 8;
        ASYNC_COPY_B128(ks + row * PADR + ch, kg + row * DQK + ch);
    }
    #pragma unroll
    for (int i = 0; i < 8; ++i) {                  // V tile: 256 rows x 32 j
        const int idx = i * 128 + tid;
        const int row = idx >> 2, ch = (idx & 3) * 8;
        ASYNC_COPY_B128(vs + row * PADR + ch, vg + (size_t)row * NN + ch);
    }
}

__global__ __launch_bounds__(128) void fa_kernel(
    const _Float16* __restrict__ Qh, const _Float16* __restrict__ Kh,
    const _Float16* __restrict__ Vh, const float* __restrict__ x,
    const float* __restrict__ gamma, float* __restrict__ out)
{
    __shared__ _Float16 Ks[2][32 * PADR];          // 2 x 2.5KB
    __shared__ _Float16 Vs[2][256 * PADR];         // 2 x 20KB

    const int tid  = threadIdx.x;
    const int b    = blockIdx.x >> 6;
    const int wave = tid >> 5;
    const int lane = tid & 31;
    const int hl   = lane >> 4;
    const int l15  = lane & 15;
    const int i0   = (blockIdx.x & 63) * 64 + wave * 16;

    const _Float16* kbase = Kh + (size_t)b * NN * DQK;
    const _Float16* vbase = Vh + (size_t)b * CC * NN;

    const v16h Bq = ld_frag(Qh + ((size_t)b * NN + i0 + l15) * DQK, hl);

    float m    = -3.0e38f;
    float lsum = 0.0f;
    v8f O[16];
    #pragma unroll
    for (int t = 0; t < 16; ++t) O[t] = v8f{};
    const v8f zero = v8f{};

    fa_stage(kbase, vbase, &Ks[0][0], &Vs[0][0], tid);   // j0 = 0

    for (int j0 = 0; j0 < NN; j0 += 32) {
        const int p = (j0 >> 5) & 1;
        ASYNC_WAIT();
        __syncthreads();
        if (j0 + 32 < NN)
            fa_stage(kbase + (size_t)(j0 + 32) * DQK, vbase + j0 + 32,
                     &Ks[p ^ 1][0], &Vs[p ^ 1][0], tid);

        const _Float16* ksp = &Ks[p][0];
        const _Float16* vsp = &Vs[p][0];

        const v16h Ak0 = ld_frag(ksp + l15 * PADR, hl);
        const v16h Ak1 = ld_frag(ksp + (16 + l15) * PADR, hl);
        v8f St0 = wmma_f16(Ak0, Bq, zero);
        v8f St1 = wmma_f16(Ak1, Bq, zero);

        float mloc = St0[0];
        #pragma unroll
        for (int t = 1; t < 8; ++t) mloc = fmaxf(mloc, St0[t]);
        #pragma unroll
        for (int t = 0; t < 8; ++t) mloc = fmaxf(mloc, St1[t]);
        mloc = fmaxf(mloc, __shfl_xor(mloc, 16, 32));
        const float mnew  = fmaxf(m, mloc);
        const float alpha = __expf(m - mnew);

        V16U Bp;                                   // P^T fragment from St regs
        float rs = 0.0f;
        #pragma unroll
        for (int t = 0; t < 8; ++t) {
            const float pv = __expf(St0[t] - mnew);
            rs += pv; Bp.h[0][t] = (_Float16)pv;
        }
        #pragma unroll
        for (int t = 0; t < 8; ++t) {
            const float pv = __expf(St1[t] - mnew);
            rs += pv; Bp.h[1][t] = (_Float16)pv;
        }
        rs += __shfl_xor(rs, 16, 32);
        lsum = lsum * alpha + rs;
        m = mnew;

        #pragma unroll
        for (int ct = 0; ct < 16; ++ct)
            #pragma unroll
            for (int t = 0; t < 8; ++t) O[ct][t] *= alpha;

        #pragma unroll
        for (int ct = 0; ct < 16; ++ct) {
            const v16h Av = ld_frag(vsp + (ct * 16 + l15) * PADR, hl);
            O[ct] = wmma_f16(Av, Bp.v, O[ct]);
        }
    }

    const float inv_l = 1.0f / lsum;
    const float g = gamma[0];
    const int n = i0 + l15;
    #pragma unroll
    for (int ct = 0; ct < 16; ++ct) {
        #pragma unroll
        for (int t = 0; t < 8; ++t) {
            const int c = ct * 16 + hl * 8 + t;
            const size_t off = ((size_t)b * CC + c) * NN + n;
            out[off] = g * (O[ct][t] * inv_l) + x[off];
        }
    }
}

// ---------------------------------------------------------------------------
extern "C" void kernel_launch(void* const* d_in, const int* in_sizes, int n_in,
                              void* d_out, int out_size, void* d_ws, size_t ws_size,
                              hipStream_t stream)
{
    (void)in_sizes; (void)n_in; (void)out_size; (void)ws_size;
    const float* x     = (const float*)d_in[0];
    const float* Wq    = (const float*)d_in[1];
    const float* bq    = (const float*)d_in[2];
    const float* Wk    = (const float*)d_in[3];
    const float* bk    = (const float*)d_in[4];
    const float* Wv    = (const float*)d_in[5];
    const float* bv    = (const float*)d_in[6];
    const float* gamma = (const float*)d_in[7];
    float* out = (float*)d_out;

    // workspace: Qh (4MB) | Kh (4MB) | Vh (32MB) | Wqkh (32KB) | Wvh (128KB)
    _Float16* Qh   = (_Float16*)d_ws;
    _Float16* Kh   = Qh + (size_t)BB * NN * DQK;
    _Float16* Vh   = Kh + (size_t)BB * NN * DQK;
    _Float16* Wqkh = Vh + (size_t)BB * CC * NN;
    _Float16* Wvh  = Wqkh + (size_t)2 * DQK * CC;

    wcvt_kernel<<<(CC * CC + 255) / 256, 256, 0, stream>>>(Wq, Wk, Wv, Wqkh, Wvh);
    proj_kernel<<<BB * (NN / 64), 128, 0, stream>>>(x, Wqkh, Wvh, bq, bk, bv,
                                                    Qh, Kh, Vh);
    fa_kernel<<<BB * (NN / 64), 128, 0, stream>>>(Qh, Kh, Vh, x, gamma, out);
}
